// Ctx_MHAtt_5918464934029
// MI455X (gfx1250) — compile-verified, matrix operand order checked by reading, not run on hardware
//
#include <hip/hip_runtime.h>
#include <hip/hip_bf16.h>
#include <cmath>

#define B_  8
#define NQ_ 1024
#define NK_ 1024
#define HD_ 1024
#define H_  8
#define HS_ 128
#define KC_ 32

typedef __attribute__((ext_vector_type(16))) _Float16 v16h;
typedef __attribute__((ext_vector_type(8)))  float    v8f;

union Frag16 { v16h v; uint4 q[2]; _Float16 e[16]; };

__device__ __forceinline__ v8f wmma_f16(v16h a, v16h b, v8f c) {
    // D = A(16x32) * B(32x16) + C, f32 accumulate
    return __builtin_amdgcn_wmma_f32_16x16x32_f16(false, a, false, b, (short)0, c, false, false);
}

// A-layout fragment load of a 16x32 f16 tile; element (r,k) at base[r*stride + k].
// Per ISA 7.12.2: lane<16 holds row M=lane, K = 0..7 and 16..23; lane>=16 holds K = 8..15 and 24..31.
// Also used to build B fragments by loading B-transpose tiles (layout duality).
__device__ __forceinline__ v16h frag_ld(const _Float16* __restrict__ base, int stride, int lane) {
    const int r  = lane & 15;
    const int kb = (lane >> 4) << 3;     // 0 or 8
    const _Float16* p = base + (size_t)r * stride + kb;
    Frag16 f;
    f.q[0] = *(const uint4*)(p);
    f.q[1] = *(const uint4*)(p + 16);
    return f.v;
}

// ---------------------------------------------------------------------------
// Projection GEMM: Y[m][n] = sum_k X[m][k]*W[n][k] + bias[n]
// X: [B][Mrows][HD] f32 ; W: [HD][HD] f32 ; out f16 in heads layout.
// OUT_T==0: out[(b*H+h)][m][hs]   (normal heads)
// OUT_T==1: out[(b*H+h)][hs][m]   (transposed heads, used for V)
// ---------------------------------------------------------------------------
template<int OUT_T>
__global__ __launch_bounds__(256)
void proj_kernel(const float* __restrict__ X, const float* __restrict__ W,
                 const float* __restrict__ bias, _Float16* __restrict__ out, int Mrows)
{
    __shared__ __align__(16) _Float16 As[128][40];
    __shared__ __align__(16) _Float16 Bs[128][40];

    const int tid  = threadIdx.x;
    const int lane = tid & 31;
    const int w    = tid >> 5;
    const int m0   = blockIdx.x * 128;
    const int n0   = blockIdx.y * 128;
    const int b    = blockIdx.z;
    const int wrow = w >> 2;   // 0..1  (64 rows each)
    const int wcol = w & 3;    // 0..3  (32 cols each)
    const int ln16 = lane & 15, hsel = lane >> 4;

    const float* Xb = X + (size_t)b * Mrows * HD_;

    v8f acc[4][2];
    #pragma unroll
    for (int mt = 0; mt < 4; ++mt)
        #pragma unroll
        for (int nt = 0; nt < 2; ++nt)
            #pragma unroll
            for (int i = 0; i < 8; ++i) acc[mt][nt][i] = 0.f;

    for (int k0 = 0; k0 < HD_; k0 += 32) {
        #pragma unroll
        for (int it = 0; it < 4; ++it) {
            int idx = tid + it * 256;       // 0..1023
            int row = idx >> 3;
            int c4  = (idx & 7) * 4;
            float4 xv = make_float4(0.f, 0.f, 0.f, 0.f);
            if (m0 + row < Mrows)
                xv = *(const float4*)(Xb + (size_t)(m0 + row) * HD_ + k0 + c4);
            As[row][c4 + 0] = (_Float16)xv.x; As[row][c4 + 1] = (_Float16)xv.y;
            As[row][c4 + 2] = (_Float16)xv.z; As[row][c4 + 3] = (_Float16)xv.w;
            float4 wv = *(const float4*)(W + (size_t)(n0 + row) * HD_ + k0 + c4);
            Bs[row][c4 + 0] = (_Float16)wv.x; Bs[row][c4 + 1] = (_Float16)wv.y;
            Bs[row][c4 + 2] = (_Float16)wv.z; Bs[row][c4 + 3] = (_Float16)wv.w;
        }
        __syncthreads();

        v16h afr[4], bfr[2];
        #pragma unroll
        for (int mt = 0; mt < 4; ++mt)
            afr[mt] = frag_ld(&As[wrow * 64 + mt * 16][0], 40, lane);
        #pragma unroll
        for (int nt = 0; nt < 2; ++nt)
            bfr[nt] = frag_ld(&Bs[wcol * 32 + nt * 16][0], 40, lane);
        #pragma unroll
        for (int mt = 0; mt < 4; ++mt)
            #pragma unroll
            for (int nt = 0; nt < 2; ++nt)
                acc[mt][nt] = wmma_f16(afr[mt], bfr[nt], acc[mt][nt]);
        __syncthreads();
    }

    // epilogue: bias + f16 store into heads layout
    #pragma unroll
    for (int mt = 0; mt < 4; ++mt) {
        #pragma unroll
        for (int nt = 0; nt < 2; ++nt) {
            int n  = n0 + wcol * 32 + nt * 16 + ln16;
            float bv = bias[n];
            int h  = n >> 7, hs = n & 127;
            if (OUT_T == 0) {
                #pragma unroll
                for (int i = 0; i < 8; ++i) {
                    int m = m0 + wrow * 64 + mt * 16 + i + 8 * hsel;
                    if (m < Mrows)
                        out[(size_t)(b * H_ + h) * Mrows * HS_ + (size_t)m * HS_ + hs] =
                            (_Float16)(acc[mt][nt][i] + bv);
                }
            } else {
                int mb = m0 + wrow * 64 + mt * 16 + 8 * hsel;   // multiple of 8, Mrows==1024 here
                __align__(16) _Float16 t8[8];
                #pragma unroll
                for (int i = 0; i < 8; ++i) t8[i] = (_Float16)(acc[mt][nt][i] + bv);
                *(uint4*)(out + (size_t)(b * H_ + h) * HS_ * Mrows + (size_t)hs * Mrows + mb) =
                    *(const uint4*)t8;
            }
        }
    }
}

// ---------------------------------------------------------------------------
// qmix[bh][u][d] = (qh[bh][u][d] + tanh(tau)*sum_c Wup[u][c]*ch[bh][c][d]) / (1+tanh(tau))
// ---------------------------------------------------------------------------
__global__ __launch_bounds__(128)
void qmix_kernel(_Float16* __restrict__ qh, const _Float16* __restrict__ ch,
                 const float* __restrict__ Wup, const float* __restrict__ tau)
{
    const int d  = threadIdx.x;      // 0..127
    const int u  = blockIdx.x;       // 0..1023
    const int bh = blockIdx.y;       // 0..63
    const float t   = tanhf(tau[0]);
    const float inv = 1.f / (1.f + t);
    const _Float16* cp = ch + (size_t)bh * KC_ * HS_ + d;
    const float*    wu = Wup + (size_t)u * KC_;
    float acc = 0.f;
    #pragma unroll
    for (int c = 0; c < KC_; ++c) acc += wu[c] * (float)cp[c * HS_];
    size_t idx = (size_t)bh * NQ_ * HS_ + (size_t)u * HS_ + d;
    qh[idx] = (_Float16)(((float)qh[idx] + t * acc) * inv);
}

// ---------------------------------------------------------------------------
// Attention: per block (bh, 16 q rows). 8 waves each own a 128-key score slice.
// Scores in registers -> layernorm(mean/var over 1024 keys) -> softmax ->
// P (unnormalized exp) staged f16 in LDS -> P @ V (full K=1024) per wave.
// ---------------------------------------------------------------------------
__global__ __launch_bounds__(256)
void attn_kernel(const _Float16* __restrict__ qm,   // [BH][NQ][HS] (qmix)
                 const _Float16* __restrict__ kh,   // [BH][NK][HS]
                 const _Float16* __restrict__ vhT,  // [BH][HS][NK]
                 const float* __restrict__ lnw, const float* __restrict__ lnb,
                 _Float16* __restrict__ attout)     // [B][NQ][H][HS]
{
    __shared__ __align__(16) _Float16 P[16][1032];
    __shared__ float redA[8][16], redB[8][16];
    __shared__ float muS[16], rsS[16], mxS[16], dnS[16];

    const int tid  = threadIdx.x;
    const int lane = tid & 31;
    const int w    = tid >> 5;
    const int q0   = blockIdx.x * 16;
    const int bh   = blockIdx.y;
    const int bq   = bh / H_, hh = bh % H_;
    const int ln16 = lane & 15, hsel = lane >> 4;
    const int key0 = w * 128;

    const _Float16* qp = qm  + (size_t)bh * NQ_ * HS_;
    const _Float16* kp = kh  + (size_t)bh * NK_ * HS_;
    const _Float16* vp = vhT + (size_t)bh * HS_ * NK_;

    // preload layernorm params for this lane's key columns
    float wv[8], bv[8];
    #pragma unroll
    for (int ct = 0; ct < 8; ++ct) {
        int kk = key0 + ct * 16 + ln16;
        wv[ct] = lnw[kk]; bv[ct] = lnb[kk];
    }

    // ---- Phase 1: S = qmix @ kh^T, 16x128 per wave -------------------------
    v8f acc[8];
    #pragma unroll
    for (int ct = 0; ct < 8; ++ct)
        #pragma unroll
        for (int i = 0; i < 8; ++i) acc[ct][i] = 0.f;

    for (int kd = 0; kd < HS_; kd += 32) {
        v16h a = frag_ld(qp + (size_t)q0 * HS_ + kd, HS_, lane);
        #pragma unroll
        for (int ct = 0; ct < 8; ++ct) {
            v16h b = frag_ld(kp + (size_t)(key0 + ct * 16) * HS_ + kd, HS_, lane);
            acc[ct] = wmma_f16(a, b, acc[ct]);
        }
    }
    const float scal = 0.08838834764831845f;   // 1/sqrt(128)
    #pragma unroll
    for (int ct = 0; ct < 8; ++ct)
        #pragma unroll
        for (int i = 0; i < 8; ++i) acc[ct][i] *= scal;

    // ---- Round 1: mean / var over keys ------------------------------------
    float ps[8], pq[8];
    #pragma unroll
    for (int i = 0; i < 8; ++i) { ps[i] = 0.f; pq[i] = 0.f; }
    #pragma unroll
    for (int ct = 0; ct < 8; ++ct)
        #pragma unroll
        for (int i = 0; i < 8; ++i) { float x = acc[ct][i]; ps[i] += x; pq[i] += x * x; }
    #pragma unroll
    for (int m = 8; m >= 1; m >>= 1)
        #pragma unroll
        for (int i = 0; i < 8; ++i) { ps[i] += __shfl_xor(ps[i], m); pq[i] += __shfl_xor(pq[i], m); }
    if (ln16 == 0) {
        #pragma unroll
        for (int i = 0; i < 8; ++i) { redA[w][hsel * 8 + i] = ps[i]; redB[w][hsel * 8 + i] = pq[i]; }
    }
    __syncthreads();
    if (tid < 16) {
        float s = 0.f, sq = 0.f;
        for (int w2 = 0; w2 < 8; ++w2) { s += redA[w2][tid]; sq += redB[w2][tid]; }
        float mu  = s  * (1.f / NK_);
        float var = sq * (1.f / NK_) - mu * mu;
        muS[tid] = mu;
        rsS[tid] = rsqrtf(var + 1e-5f);
    }
    __syncthreads();

    // ---- Round 2: apply LN, row max ---------------------------------------
    float pm[8];
    #pragma unroll
    for (int i = 0; i < 8; ++i) pm[i] = -1e30f;
    #pragma unroll
    for (int ct = 0; ct < 8; ++ct)
        #pragma unroll
        for (int i = 0; i < 8; ++i) {
            int m = i + 8 * hsel;
            float x = (acc[ct][i] - muS[m]) * rsS[m] * wv[ct] + bv[ct];
            acc[ct][i] = x;
            pm[i] = fmaxf(pm[i], x);
        }
    #pragma unroll
    for (int m = 8; m >= 1; m >>= 1)
        #pragma unroll
        for (int i = 0; i < 8; ++i) pm[i] = fmaxf(pm[i], __shfl_xor(pm[i], m));
    if (ln16 == 0) {
        #pragma unroll
        for (int i = 0; i < 8; ++i) redA[w][hsel * 8 + i] = pm[i];
    }
    __syncthreads();
    if (tid < 16) {
        float mx = -1e30f;
        for (int w2 = 0; w2 < 8; ++w2) mx = fmaxf(mx, redA[w2][tid]);
        mxS[tid] = mx;
    }
    __syncthreads();

    // ---- Round 3: exp, stage P to LDS, denom ------------------------------
    #pragma unroll
    for (int i = 0; i < 8; ++i) ps[i] = 0.f;
    #pragma unroll
    for (int ct = 0; ct < 8; ++ct)
        #pragma unroll
        for (int i = 0; i < 8; ++i) {
            int m = i + 8 * hsel;
            float e = __expf(acc[ct][i] - mxS[m]);
            ps[i] += e;
            P[m][key0 + ct * 16 + ln16] = (_Float16)e;
        }
    #pragma unroll
    for (int m = 8; m >= 1; m >>= 1)
        #pragma unroll
        for (int i = 0; i < 8; ++i) ps[i] += __shfl_xor(ps[i], m);
    if (ln16 == 0) {
        #pragma unroll
        for (int i = 0; i < 8; ++i) redA[w][hsel * 8 + i] = ps[i];
    }
    __syncthreads();
    if (tid < 16) {
        float s = 0.f;
        for (int w2 = 0; w2 < 8; ++w2) s += redA[w2][tid];
        dnS[tid] = 1.f / s;
    }
    __syncthreads();

    // ---- Phase 3: O = P @ V; wave w owns d-columns [16w, 16w+16) ----------
    const int d0 = w * 16;
    v8f oacc;
    #pragma unroll
    for (int i = 0; i < 8; ++i) oacc[i] = 0.f;
    for (int kk = 0; kk < NK_; kk += 32) {
        v16h a = frag_ld(&P[0][kk], 1032, lane);
        v16h b = frag_ld(vp + (size_t)d0 * NK_ + kk, NK_, lane);
        oacc = wmma_f16(a, b, oacc);
    }
    #pragma unroll
    for (int i = 0; i < 8; ++i) {
        int m = i + 8 * hsel;
        float o = oacc[i] * dnS[m];
        size_t idx = (((size_t)(bq * NQ_ + q0 + m)) * H_ + hh) * HS_ + (d0 + ln16);
        attout[idx] = (_Float16)o;
    }
}

// ---------------------------------------------------------------------------
// Final projection: out[m][n] = sum_k Xh[m][k]*W[n][k] + bias[n]  (f16 in, f32 out)
// Xh: [B*NQ][HD] f16, out: [B*NQ][HD] f32
// ---------------------------------------------------------------------------
__global__ __launch_bounds__(256)
void final_proj_kernel(const _Float16* __restrict__ Xh, const float* __restrict__ W,
                       const float* __restrict__ bias, float* __restrict__ out)
{
    __shared__ __align__(16) _Float16 As[128][40];
    __shared__ __align__(16) _Float16 Bs[128][40];

    const int tid  = threadIdx.x;
    const int lane = tid & 31;
    const int w    = tid >> 5;
    const int m0   = blockIdx.x * 128;
    const int n0   = blockIdx.y * 128;
    const int wrow = w >> 2, wcol = w & 3;
    const int ln16 = lane & 15, hsel = lane >> 4;

    v8f acc[4][2];
    #pragma unroll
    for (int mt = 0; mt < 4; ++mt)
        #pragma unroll
        for (int nt = 0; nt < 2; ++nt)
            #pragma unroll
            for (int i = 0; i < 8; ++i) acc[mt][nt][i] = 0.f;

    for (int k0 = 0; k0 < HD_; k0 += 32) {
        #pragma unroll
        for (int it = 0; it < 2; ++it) {
            int idx = tid + it * 256;       // 0..511
            int row = idx >> 2;
            int c8  = (idx & 3) * 8;
            *(uint4*)&As[row][c8] = *(const uint4*)(Xh + (size_t)(m0 + row) * HD_ + k0 + c8);
        }
        #pragma unroll
        for (int it = 0; it < 4; ++it) {
            int idx = tid + it * 256;
            int row = idx >> 3;
            int c4  = (idx & 7) * 4;
            float4 wvv = *(const float4*)(W + (size_t)(n0 + row) * HD_ + k0 + c4);
            Bs[row][c4 + 0] = (_Float16)wvv.x; Bs[row][c4 + 1] = (_Float16)wvv.y;
            Bs[row][c4 + 2] = (_Float16)wvv.z; Bs[row][c4 + 3] = (_Float16)wvv.w;
        }
        __syncthreads();

        v16h afr[4], bfr[2];
        #pragma unroll
        for (int mt = 0; mt < 4; ++mt)
            afr[mt] = frag_ld(&As[wrow * 64 + mt * 16][0], 40, lane);
        #pragma unroll
        for (int nt = 0; nt < 2; ++nt)
            bfr[nt] = frag_ld(&Bs[wcol * 32 + nt * 16][0], 40, lane);
        #pragma unroll
        for (int mt = 0; mt < 4; ++mt)
            #pragma unroll
            for (int nt = 0; nt < 2; ++nt)
                acc[mt][nt] = wmma_f16(afr[mt], bfr[nt], acc[mt][nt]);
        __syncthreads();
    }

    #pragma unroll
    for (int mt = 0; mt < 4; ++mt)
        #pragma unroll
        for (int nt = 0; nt < 2; ++nt) {
            int n = n0 + wcol * 32 + nt * 16 + ln16;
            float bvv = bias[n];
            #pragma unroll
            for (int i = 0; i < 8; ++i) {
                int m = m0 + wrow * 64 + mt * 16 + i + 8 * hsel;
                out[(size_t)m * HD_ + n] = acc[mt][nt][i] + bvv;
            }
        }
}

// ---------------------------------------------------------------------------
extern "C" void kernel_launch(void* const* d_in, const int* in_sizes, int n_in,
                              void* d_out, int out_size, void* d_ws, size_t ws_size,
                              hipStream_t stream)
{
    (void)in_sizes; (void)n_in; (void)out_size; (void)ws_size;

    const float* v   = (const float*)d_in[0];
    const float* k   = (const float*)d_in[1];
    const float* q   = (const float*)d_in[2];
    const float* c   = (const float*)d_in[3];
    const float* Wv  = (const float*)d_in[4];
    const float* bv  = (const float*)d_in[5];
    const float* Wk  = (const float*)d_in[6];
    const float* bk  = (const float*)d_in[7];
    const float* Wq  = (const float*)d_in[8];
    const float* bq  = (const float*)d_in[9];
    const float* Wc  = (const float*)d_in[10];
    const float* bc  = (const float*)d_in[11];
    const float* Wm  = (const float*)d_in[12];
    const float* bm  = (const float*)d_in[13];
    const float* Wup = (const float*)d_in[14];
    const float* lnw = (const float*)d_in[15];
    const float* lnb = (const float*)d_in[16];
    const float* tau = (const float*)d_in[17];

    // workspace partition (f16 elements); total ~64.5 MB
    _Float16* ws  = (_Float16*)d_ws;
    const size_t NBHT = (size_t)B_ * H_ * NK_ * HS_;   // 8,388,608
    _Float16* vhT = ws;
    _Float16* kh  = vhT + NBHT;
    _Float16* qh  = kh  + NBHT;
    _Float16* ch  = qh  + NBHT;                        // 262,144 used
    _Float16* att = ch  + (size_t)B_ * H_ * KC_ * HS_;

    dim3 blk(256);
    proj_kernel<1><<<dim3(8, 8, B_), blk, 0, stream>>>(v, Wv, bv, vhT, NK_);   // V, transposed heads
    proj_kernel<0><<<dim3(8, 8, B_), blk, 0, stream>>>(k, Wk, bk, kh,  NK_);   // K
    proj_kernel<0><<<dim3(8, 8, B_), blk, 0, stream>>>(q, Wq, bq, qh,  NQ_);   // Q
    proj_kernel<0><<<dim3(1, 8, B_), blk, 0, stream>>>(c, Wc, bc, ch,  KC_);   // C (32 rows)

    qmix_kernel<<<dim3(NQ_, B_ * H_), dim3(128), 0, stream>>>(qh, ch, Wup, tau);

    attn_kernel<<<dim3(NQ_ / 16, B_ * H_), blk, 0, stream>>>(qh, kh, vhT, lnw, lnb, att);

    final_proj_kernel<<<dim3((B_ * NQ_) / 128, 8), blk, 0, stream>>>(att, Wm, bm, (float*)d_out);
}
